// GAT_74354473828959
// MI455X (gfx1250) — matrix operand order repaired and checked by reading, not compiled, not run on hardware
//
#include <hip/hip_runtime.h>
#include <hip/hip_bf16.h>

// ---------------------------------------------------------------------------
// GAT (3x GATConv + BN/ReLU) for MI455X / gfx1250.
// GEMMs via v_wmma_f32_16x16x32_bf16 (wave32, one wave per 16-row tile).
// Edge softmax/aggregation via L2-resident float atomics (23.3 TB/s HBM,
// 192MB L2 holds all per-node tables).
// ---------------------------------------------------------------------------

#define GAT_N 100000
#define GAT_E 1600000
#define GAT_SLOPE 0.2f

typedef __bf16 v16bf __attribute__((ext_vector_type(16)));
typedef float  v8f   __attribute__((ext_vector_type(8)));

// ---------------- dense feature transform: H = X @ W (bf16 WMMA) -----------
// One wave (32 lanes) per 16-row tile; F/16 column tiles accumulated in v8f.
// A-fragment layout (16-bit A 16x32, ISA 7.12.2):
//   lane<16 : M=lane,    K = {k0+0..7, k0+16..23} packed 2-per-VGPR
//   lane>=16: M=lane-16, K = {k0+8..15, k0+24..31}
// B-fragment layout (32x16): same K order as A (lane-half h: K = {k0+8h..8h+7,
//   k0+16+8h..16+8h+7}), lane&15 = column, 2 consecutive K per VGPR.
template <int K, int F>
__global__ __launch_bounds__(32) void gemm_bf16_wmma(
    const float* __restrict__ X, const float* __restrict__ W,
    float* __restrict__ H) {
  constexpr int FT = F / 16;
  const int lane = threadIdx.x & 31;
  const int half = lane >> 4;
  const int l16  = lane & 15;
  const int m0   = blockIdx.x * 16;
  const int mrow = m0 + l16;

  v8f acc[FT] = {};

#pragma unroll
  for (int k0 = 0; k0 < K; k0 += 32) {
    v16bf a;
#pragma unroll
    for (int r = 0; r < 8; ++r) {
      const int kb = k0 + ((r < 4) ? 0 : 16) + half * 8 + 2 * (r & 3);
      const float2 xv = *(const float2*)(X + (size_t)mrow * K + kb);
      a[2 * r + 0] = (__bf16)xv.x;
      a[2 * r + 1] = (__bf16)xv.y;
    }
#pragma unroll
    for (int t = 0; t < FT; ++t) {
      v16bf b;
      const int col = t * 16 + l16;
#pragma unroll
      for (int r = 0; r < 8; ++r) {
        const int k = k0 + ((r < 4) ? 0 : 16) + half * 8 + 2 * (r & 3);
        b[2 * r + 0] = (__bf16)W[(size_t)k * F + col];
        b[2 * r + 1] = (__bf16)W[(size_t)(k + 1) * F + col];
      }
      acc[t] = __builtin_amdgcn_wmma_f32_16x16x32_bf16(
          false, a, false, b, (short)0, acc[t], false, false);
    }
  }
  // D layout: VGPR r -> row m0 + half*8 + r, col = t*16 + (lane&15)
#pragma unroll
  for (int t = 0; t < FT; ++t)
#pragma unroll
    for (int r = 0; r < 8; ++r)
      H[(size_t)(m0 + half * 8 + r) * F + t * 16 + l16] = acc[t][r];
}

// ---------------- per-node attention scores: ss = H.a_s, sd = H.a_d --------
template <int F>
__global__ void node_scores(const float* __restrict__ H,
                            const float* __restrict__ a_s,
                            const float* __restrict__ a_d,
                            float* __restrict__ ss, float* __restrict__ sd,
                            int n) {
  const int wid  = (int)((blockIdx.x * (size_t)blockDim.x + threadIdx.x) >> 5);
  const int lane = threadIdx.x & 31;
  if (wid >= n) return;
  float s1 = 0.f, s2 = 0.f;
#pragma unroll
  for (int f = lane; f < F; f += 32) {
    const float h = H[(size_t)wid * F + f];
    s1 += h * a_s[f];
    s2 += h * a_d[f];
  }
#pragma unroll
  for (int off = 16; off; off >>= 1) {
    s1 += __shfl_xor(s1, off);
    s2 += __shfl_xor(s2, off);
  }
  if (lane == 0) { ss[wid] = s1; sd[wid] = s2; }
}

// ---------------- init: segment max/sum seeds + bias-broadcast output ------
__global__ void init_layer(float* __restrict__ out, const float* __restrict__ b,
                           float* __restrict__ m, float* __restrict__ s,
                           int n, int F) {
  const int i = blockIdx.x * blockDim.x + threadIdx.x;
  if (i < n) { m[i] = -__builtin_inff(); s[i] = 0.f; }
  if (i < n * F) out[i] = b[i % F];
}

// float atomic max via sign-split integer atomics (monotone encoding).
__device__ __forceinline__ void atomicMaxF(float* addr, float val) {
  if (val >= 0.f)
    atomicMax((int*)addr, __float_as_int(val));
  else
    atomicMin((unsigned int*)addr, (unsigned int)__float_as_int(val));
}

// ---------------- edge pass 1: leaky-relu score + segment max --------------
__global__ void edge_max(const int* __restrict__ ei,
                         const float* __restrict__ ss,
                         const float* __restrict__ sd,
                         float* __restrict__ ev, float* __restrict__ m) {
  const int i = blockIdx.x * blockDim.x + threadIdx.x;
  const int Et = GAT_E + GAT_N;
  if (i >= Et) return;
  const int s = (i < GAT_E) ? ei[i]         : (i - GAT_E);  // self loops
  const int d = (i < GAT_E) ? ei[GAT_E + i] : (i - GAT_E);
  float e = ss[s] + sd[d];
  e = (e > 0.f) ? e : GAT_SLOPE * e;
  ev[i] = e;
  atomicMaxF(m + d, e);
}

// ---------------- edge pass 2: p = exp(e - m[dst]), segment sum ------------
__global__ void edge_expsum(const int* __restrict__ ei,
                            float* __restrict__ ev,
                            const float* __restrict__ m,
                            float* __restrict__ s) {
  const int i = blockIdx.x * blockDim.x + threadIdx.x;
  const int Et = GAT_E + GAT_N;
  if (i >= Et) return;
  const int d = (i < GAT_E) ? ei[GAT_E + i] : (i - GAT_E);
  const float p = __expf(ev[i] - m[d]);
  ev[i] = p;
  atomicAdd(s + d, p);
}

// ---------------- edge pass 3: out[dst] += alpha * h[src] ------------------
template <int F>
__global__ void edge_scatter(const int* __restrict__ ei,
                             const float* __restrict__ H,
                             const float* __restrict__ ev,
                             const float* __restrict__ ssum,
                             float* __restrict__ out) {
  const long long t = (long long)blockIdx.x * blockDim.x + threadIdx.x;
  const int edge = (int)(t / F);
  const int f    = (int)(t % F);
  const int Et   = GAT_E + GAT_N;
  if (edge >= Et) return;
  const int s = (edge < GAT_E) ? ei[edge]         : (edge - GAT_E);
  const int d = (edge < GAT_E) ? ei[GAT_E + edge] : (edge - GAT_E);
  const float alpha = ev[edge] / ssum[d];
  atomicAdd(out + (size_t)d * F + f, H[(size_t)s * F + f] * alpha);
}

// ---------------- BatchNorm (eval) + ReLU, in place ------------------------
__global__ void bn_relu(float* __restrict__ x, const float* __restrict__ g,
                        const float* __restrict__ be,
                        const float* __restrict__ rm,
                        const float* __restrict__ rv, int n, int F) {
  const int i = blockIdx.x * blockDim.x + threadIdx.x;
  if (i >= n * F) return;
  const int f = i % F;
  const float v = (x[i] - rm[f]) * rsqrtf(rv[f] + 1e-5f) * g[f] + be[f];
  x[i] = v > 0.f ? v : 0.f;
}

// ---------------------------------------------------------------------------
extern "C" void kernel_launch(void* const* d_in, const int* in_sizes, int n_in,
                              void* d_out, int out_size, void* d_ws,
                              size_t ws_size, hipStream_t stream) {
  (void)in_sizes; (void)n_in; (void)out_size; (void)ws_size;
  const float* x   = (const float*)d_in[0];
  const int*   ei  = (const int*)d_in[1];
  const float* W1  = (const float*)d_in[2];
  const float* as1 = (const float*)d_in[3];
  const float* ad1 = (const float*)d_in[4];
  const float* b1  = (const float*)d_in[5];
  const float* g1  = (const float*)d_in[6];
  const float* be1 = (const float*)d_in[7];
  const float* rm1 = (const float*)d_in[8];
  const float* rv1 = (const float*)d_in[9];
  const float* W2  = (const float*)d_in[10];
  const float* as2 = (const float*)d_in[11];
  const float* ad2 = (const float*)d_in[12];
  const float* b2  = (const float*)d_in[13];
  const float* g2  = (const float*)d_in[14];
  const float* be2 = (const float*)d_in[15];
  const float* rm2 = (const float*)d_in[16];
  const float* rv2 = (const float*)d_in[17];
  const float* W3  = (const float*)d_in[18];
  const float* as3 = (const float*)d_in[19];
  const float* ad3 = (const float*)d_in[20];
  const float* b3  = (const float*)d_in[21];
  float* out = (float*)d_out;

  const int N = GAT_N, Et = GAT_E + GAT_N;
  float* ws   = (float*)d_ws;
  float* bufA = ws;               // N*64 : h of current layer
  float* bufB = bufA + (size_t)N * 64;  // N*64 : aggregated out / next x
  float* ss   = bufB + (size_t)N * 64;  // N
  float* sd   = ss + N;                 // N
  float* mx   = sd + N;                 // N (segment max)
  float* sm   = mx + N;                 // N (segment sum)
  float* ev   = sm + N;                 // Et (edge score -> edge prob)

  const int eB = (Et + 255) / 256;
  const int nwB = (N * 32 + 255) / 256;   // node_scores: 1 wave/node
  const int nf64 = (N * 64 + 255) / 256;
  const int nf32 = (N * 32 + 255) / 256;
  const int sc64 = (int)(((long long)Et * 64 + 255) / 256);
  const int sc32 = (int)(((long long)Et * 32 + 255) / 256);

  // ---------------- layer 1: GATConv(128->64) + BN + ReLU ----------------
  gemm_bf16_wmma<128, 64><<<N / 16, 32, 0, stream>>>(x, W1, bufA);
  node_scores<64><<<nwB, 256, 0, stream>>>(bufA, as1, ad1, ss, sd, N);
  init_layer<<<nf64, 256, 0, stream>>>(bufB, b1, mx, sm, N, 64);
  edge_max<<<eB, 256, 0, stream>>>(ei, ss, sd, ev, mx);
  edge_expsum<<<eB, 256, 0, stream>>>(ei, ev, mx, sm);
  edge_scatter<64><<<sc64, 256, 0, stream>>>(ei, bufA, ev, sm, bufB);
  bn_relu<<<nf64, 256, 0, stream>>>(bufB, g1, be1, rm1, rv1, N, 64);

  // ---------------- layer 2: GATConv(64->64) + BN + ReLU -----------------
  gemm_bf16_wmma<64, 64><<<N / 16, 32, 0, stream>>>(bufB, W2, bufA);
  node_scores<64><<<nwB, 256, 0, stream>>>(bufA, as2, ad2, ss, sd, N);
  init_layer<<<nf64, 256, 0, stream>>>(bufB, b2, mx, sm, N, 64);  // bufB reused
  edge_max<<<eB, 256, 0, stream>>>(ei, ss, sd, ev, mx);
  edge_expsum<<<eB, 256, 0, stream>>>(ei, ev, mx, sm);
  edge_scatter<64><<<sc64, 256, 0, stream>>>(ei, bufA, ev, sm, bufB);
  bn_relu<<<nf64, 256, 0, stream>>>(bufB, g2, be2, rm2, rv2, N, 64);

  // ---------------- layer 3: GATConv(64->32) -> d_out --------------------
  gemm_bf16_wmma<64, 32><<<N / 16, 32, 0, stream>>>(bufB, W3, bufA);
  node_scores<32><<<nwB, 256, 0, stream>>>(bufA, as3, ad3, ss, sd, N);
  init_layer<<<nf32, 256, 0, stream>>>(out, b3, mx, sm, N, 32);
  edge_max<<<eB, 256, 0, stream>>>(ei, ss, sd, ev, mx);
  edge_expsum<<<eB, 256, 0, stream>>>(ei, ev, mx, sm);
  edge_scatter<32><<<sc32, 256, 0, stream>>>(ei, bufA, ev, sm, out);
}
